// QuantLinear_5634997092956
// MI455X (gfx1250) — compile-verified, matrix operand order checked by reading, not run on hardware
//
#include <hip/hip_runtime.h>
#include <hip/hip_bf16.h>

typedef __attribute__((ext_vector_type(16))) _Float16 v16h;
typedef __attribute__((ext_vector_type(2)))  _Float16 h2;
typedef __attribute__((ext_vector_type(8)))  float    v8f;
typedef __attribute__((ext_vector_type(4)))  unsigned v4u;
typedef __attribute__((ext_vector_type(8)))  int      v8i;
typedef __attribute__((ext_vector_type(4)))  int      v4i;

#define IN_F   4096
#define OUT_F  4096
#define M_TOK  4096           // 2*2048
#define NG     32             // GPTQ groups of 128
#define RNK    16
#define SCALING 2.0f

#define TM 128
#define TN 128
#define TK 32
#define XPITCH 36             // dwords per x-tile LDS row: 32 + TDM pad of 4 dwords

// ---------------------------------------------------------------------------
// Tensor Data Mover: issue a 2-D tile load (global -> LDS). D# built per
// CDNA5 ISA. All dims/strides in data_size units.
// This toolchain's builtin is the 6-arg form (clang-23 / therock headers).
__device__ __forceinline__ void tdm_load_2d(unsigned lds_addr, const void* gptr,
                                            unsigned tdim0, unsigned tdim1,
                                            unsigned tile0, unsigned tile1,
                                            unsigned stride0, unsigned flags) {
    unsigned long long ga = (unsigned long long)gptr;
    v4u g0;
    g0[0] = 1u;                                            // count=1, user mode
    g0[1] = lds_addr;                                      // LDS byte address
    g0[2] = (unsigned)(ga & 0xFFFFFFFFull);                // global_addr[31:0]
    g0[3] = (unsigned)((ga >> 32) & 0x1FFFFFFull)          // global_addr[56:32]
          | (2u << 30);                                    // type = 2 ("image")
    v8i g1;
    g1[0] = (int)flags;                                    // mask=0|data_size|pad cfg
    g1[1] = (int)((tdim0 & 0xFFFFu) << 16);                // tensor_dim0[15:0] @ b63:48
    g1[2] = (int)((tdim0 >> 16) | ((tdim1 & 0xFFFFu) << 16));
    g1[3] = (int)((tdim1 >> 16) | (tile0 << 16));          // tile_dim0 @ b127:112
    g1[4] = (int)tile1;                                    // tile_dim1, tile_dim2=0
    g1[5] = (int)stride0;                                  // tensor_dim0_stride[31:0]
    g1[6] = 0;                                             // stride hi, dim1_stride=0
    g1[7] = 0;
    v4i z4 = {0, 0, 0, 0};                                 // groups 2/3 unused (2-D)
    v8i z8 = {0, 0, 0, 0, 0, 0, 0, 0};
    __builtin_amdgcn_tensor_load_to_lds(g0, g1, z4, z4, z8, 0);
}

#define TDM_FLAGS_X  ((2u << 16) | (1u << 20) | (4u << 22) | (3u << 25))
//        data_size=4B ^        pad_en ^   every 32 dw ^   pad 4 dwords
#define TDM_FLAGS_Q  (2u << 16)   // data_size=4B, no pad

// ---------------------------------------------------------------------------
// 1) partial column sums of |x| : grid (16 col-groups, 8 token-chunks)
__global__ __launch_bounds__(256) void colabs_kernel(const float* __restrict__ x,
                                                     float* __restrict__ colsum) {
    int col = blockIdx.x * 256 + threadIdx.x;
    int row0 = blockIdx.y * (M_TOK / 8);
    float acc = 0.f;
    const float* p = x + (size_t)row0 * IN_F + col;
    #pragma unroll 4
    for (int r = 0; r < M_TOK / 8; ++r) acc += fabsf(p[(size_t)r * IN_F]);
    colsum[blockIdx.y * IN_F + col] = acc;
}

// ---------------------------------------------------------------------------
// 2) single-block reduce -> normalization scalar
__global__ __launch_bounds__(256) void stats_kernel(const float* __restrict__ colsum,
                                                    float* __restrict__ stats) {
    __shared__ float red[256];
    float acc = 0.f;
    for (int i = threadIdx.x; i < 8 * IN_F; i += 256) acc += colsum[i];
    red[threadIdx.x] = acc;
    __syncthreads();
    for (int s = 128; s > 0; s >>= 1) {
        if (threadIdx.x < s) red[threadIdx.x] += red[threadIdx.x + s];
        __syncthreads();
    }
    if (threadIdx.x == 0) {
        float mean_w = red[0] / ((float)M_TOK * (float)IN_F);
        stats[0] = 1.0f / (mean_w + 1e-6f);
    }
}

// ---------------------------------------------------------------------------
// 3) Asw[k][r] = aw[k] * (A0[r][k] + A1[r][k])
__global__ __launch_bounds__(256) void asw_kernel(const float* __restrict__ colsum,
                                                  const float* __restrict__ stats,
                                                  const float* __restrict__ lora_A,
                                                  float* __restrict__ Asw) {
    int k = blockIdx.x * 256 + threadIdx.x;
    float s = 0.f;
    #pragma unroll
    for (int c = 0; c < 8; ++c) s += colsum[c * IN_F + k];
    float aw = s * (1.0f / (float)M_TOK) * stats[0];
    #pragma unroll
    for (int r = 0; r < RNK; ++r) {
        float a = lora_A[r * IN_F + k] + lora_A[RNK * IN_F + r * IN_F + k];
        Asw[k * RNK + r] = aw * a;
    }
}

// ---------------------------------------------------------------------------
// 4) h[m][r] = sum_k x[m][k] * Asw[k][r]   (wave per row, wave32 shfl reduce)
__global__ __launch_bounds__(256) void h_kernel(const float* __restrict__ x,
                                                const float* __restrict__ Asw,
                                                float* __restrict__ h) {
    int wave = threadIdx.x >> 5, lane = threadIdx.x & 31;
    int m = blockIdx.x * 8 + wave;
    int r = lane & 15, par = lane >> 4;
    const float* xr = x + (size_t)m * IN_F;
    float acc = 0.f;
    for (int k = par; k < IN_F; k += 2)
        acc += xr[k] * Asw[k * RNK + r];
    acc += __shfl_xor(acc, 16, 32);
    if (par == 0) h[m * RNK + r] = acc;
}

// ---------------------------------------------------------------------------
// 5) main GEMM: TDM double-buffered operand staging, f16 WMMA, packed-f16
//    fast int4 dequant, rank-16 LoRA epilogue.
__global__ __launch_bounds__(256) void gemm_kernel(const float* __restrict__ x,
                                                   const int*   __restrict__ qweight,
                                                   const int*   __restrict__ qzeros,
                                                   const float* __restrict__ scales,
                                                   const float* __restrict__ bias,
                                                   const float* __restrict__ lora_B,
                                                   const float* __restrict__ h,
                                                   float*       __restrict__ out) {
    __shared__ float    Xs[2][TM * XPITCH];   // fp32 x tile, TDM-padded rows
    __shared__ unsigned Qs[2][4 * TN];        // packed int4 tile: 4 k-words x 128 n
    __shared__ float    hS[TM * RNK];
    __shared__ float    bS[TN * RNK];

    const int tid  = threadIdx.x;
    const int bn   = blockIdx.x, bm = blockIdx.y;
    const int wave = tid >> 5,  lane = tid & 31;
    const int wm   = wave >> 2, wn   = wave & 3;     // 2 x 4 wave grid
    const int l16  = lane & 15, lhalf = lane >> 4;

    // ---- prologue: stage h tile and Bsum tile (each 2048 floats, contiguous)
    {
        const float4* hg = (const float4*)(h + (size_t)bm * TM * RNK);
        const float4* b0 = (const float4*)(lora_B + (size_t)bn * TN * RNK);
        const float4* b1 = (const float4*)(lora_B + (size_t)OUT_F * RNK + (size_t)bn * TN * RNK);
        #pragma unroll
        for (int i = 0; i < 2; ++i) {
            int idx = tid + i * 256;
            ((float4*)hS)[idx] = hg[idx];
            float4 u = b0[idx], v = b1[idx];
            float4 w; w.x = u.x + v.x; w.y = u.y + v.y; w.z = u.z + v.z; w.w = u.w + v.w;
            ((float4*)bS)[idx] = w;
        }
    }

    // ---- TDM preload of tile kk=0 into buffer 0 (one wave drives the DMA)
    if (wave == 0) {
        tdm_load_2d((unsigned)(size_t)&Xs[0][0],
                    x + (size_t)bm * TM * IN_F,
                    IN_F, M_TOK, TK, TM, IN_F, TDM_FLAGS_X);
        tdm_load_2d((unsigned)(size_t)&Qs[0][0],
                    qweight + (size_t)0 * OUT_F + bn * TN,
                    OUT_F, IN_F / 8, TN, 4, OUT_F, TDM_FLAGS_Q);
        __builtin_amdgcn_s_wait_tensorcnt(0);
    }
    __syncthreads();

    v8f acc[4][2] = {};
    int cur = 0;

    for (int g = 0; g < NG; ++g) {
        // GPTQ group constants per lane-column: scale and fused f16 fma coeffs.
        // Trick: (nibble | 0x6400) as f16 == 1024 + w exactly, so
        //        w_dq = s*(1024+w) - s*(1024+z) = fma(f16(1024+w), hs, hoff).
        h2 hsv[2], hov[2];
        #pragma unroll
        for (int j = 0; j < 2; ++j) {
            int n_g = bn * TN + wn * 32 + j * 16 + l16;
            float s = scales[(size_t)g * OUT_F + n_g];
            float z = (float)(((unsigned)qzeros[(size_t)g * (OUT_F / 8) + (n_g >> 3)]
                               >> (4 * (n_g & 7))) & 0xFu);
            float off = -s * (1024.0f + z);
            hsv[j][0] = (_Float16)s;   hsv[j][1] = (_Float16)s;
            hov[j][0] = (_Float16)off; hov[j][1] = (_Float16)off;
        }
        for (int t = 0; t < 4; ++t) {
            const int kk  = g * 128 + t * TK;
            const int nxt = cur ^ 1;

            // issue next tile's DMA while this tile computes
            if (wave == 0 && kk + TK < IN_F) {
                const int kn = kk + TK;
                tdm_load_2d((unsigned)(size_t)&Xs[nxt][0],
                            x + (size_t)bm * TM * IN_F + kn,
                            IN_F, M_TOK, TK, TM, IN_F, TDM_FLAGS_X);
                tdm_load_2d((unsigned)(size_t)&Qs[nxt][0],
                            qweight + (size_t)(kn >> 3) * OUT_F + bn * TN,
                            OUT_F, IN_F / 8, TN, 4, OUT_F, TDM_FLAGS_Q);
            }

            // ---- A fragments: fp32 tile -> f16 (16-bit A lane layout:
            //      lanes 0-15 K 0-7 / 16-23, lanes 16-31 K 8-15 / 24-31)
            union Fh { v16h v; _Float16 e[16]; };
            Fh a[4], b[2];
            #pragma unroll
            for (int i = 0; i < 4; ++i) {
                const float* xp = &Xs[cur][(wm * 64 + i * 16 + l16) * XPITCH + lhalf * 8];
                float4 f0 = *(const float4*)xp;
                float4 f1 = *(const float4*)(xp + 4);
                float4 f2 = *(const float4*)(xp + 16);
                float4 f3 = *(const float4*)(xp + 20);
                a[i].e[0] = (_Float16)f0.x; a[i].e[1] = (_Float16)f0.y;
                a[i].e[2] = (_Float16)f0.z; a[i].e[3] = (_Float16)f0.w;
                a[i].e[4] = (_Float16)f1.x; a[i].e[5] = (_Float16)f1.y;
                a[i].e[6] = (_Float16)f1.z; a[i].e[7] = (_Float16)f1.w;
                a[i].e[8]  = (_Float16)f2.x; a[i].e[9]  = (_Float16)f2.y;
                a[i].e[10] = (_Float16)f2.z; a[i].e[11] = (_Float16)f2.w;
                a[i].e[12] = (_Float16)f3.x; a[i].e[13] = (_Float16)f3.y;
                a[i].e[14] = (_Float16)f3.z; a[i].e[15] = (_Float16)f3.w;
            }
            // ---- B fragments: 2 packed words per lane, v_pk_fma_f16 dequant.
            //      Nibble p pairs with p+4 in {lo16,hi16} of each masked word.
            #pragma unroll
            for (int j = 0; j < 2; ++j) {
                int n_l = wn * 32 + j * 16 + l16;
                unsigned q0 = Qs[cur][lhalf * TN + n_l];        // K kk+lhalf*8..+7
                unsigned q1 = Qs[cur][(2 + lhalf) * TN + n_l];  // K kk+16+lhalf*8..
                union H2U { unsigned u; h2 hv; };
                #pragma unroll
                for (int p = 0; p < 4; ++p) {
                    H2U t0; t0.u = ((q0 >> (4 * p)) & 0x000F000Fu) | 0x64006400u;
                    H2U t1; t1.u = ((q1 >> (4 * p)) & 0x000F000Fu) | 0x64006400u;
                    h2 r0 = t0.hv * hsv[j] + hov[j];            // v_pk_fma_f16
                    h2 r1 = t1.hv * hsv[j] + hov[j];
                    b[j].e[p]      = r0[0];                     // K = p
                    b[j].e[p + 4]  = r0[1];                     // K = p+4
                    b[j].e[8 + p]  = r1[0];                     // K = 16+p
                    b[j].e[12 + p] = r1[1];                     // K = 16+p+4
                }
            }
            #pragma unroll
            for (int i = 0; i < 4; ++i)
                #pragma unroll
                for (int j = 0; j < 2; ++j)
                    acc[i][j] = __builtin_amdgcn_wmma_f32_16x16x32_f16(
                        false, a[i].v, false, b[j].v, (short)0, acc[i][j], false, false);

            if (wave == 0 && kk + TK < IN_F)
                __builtin_amdgcn_s_wait_tensorcnt(0);   // next tile landed
            __syncthreads();                            // publish to all waves
            cur = nxt;
        }
    }

    // ---- epilogue: + bias + SCALING * <h[m,:], Bsum[n,:]>
    #pragma unroll
    for (int j = 0; j < 2; ++j) {
        int n_l = wn * 32 + j * 16 + l16;
        int n_g = bn * TN + n_l;
        float bv = bias[n_g];
        float bsr[RNK];
        #pragma unroll
        for (int q = 0; q < 4; ++q) {
            float4 tq = ((const float4*)&bS[n_l * RNK])[q];
            bsr[q * 4 + 0] = tq.x; bsr[q * 4 + 1] = tq.y;
            bsr[q * 4 + 2] = tq.z; bsr[q * 4 + 3] = tq.w;
        }
        #pragma unroll
        for (int i = 0; i < 4; ++i) {
            int m_base = wm * 64 + i * 16 + lhalf * 8;
            #pragma unroll
            for (int v = 0; v < 8; ++v) {
                int mm = m_base + v;
                const float* hr = &hS[mm * RNK];
                float dot = 0.f;
                #pragma unroll
                for (int r = 0; r < RNK; ++r) dot = fmaf(hr[r], bsr[r], dot);
                out[(size_t)(bm * TM + mm) * OUT_F + n_g] = acc[i][j][v] + bv + SCALING * dot;
            }
        }
    }
}

// ---------------------------------------------------------------------------
extern "C" void kernel_launch(void* const* d_in, const int* in_sizes, int n_in,
                              void* d_out, int out_size, void* d_ws, size_t ws_size,
                              hipStream_t stream) {
    const float* x      = (const float*)d_in[0];
    const float* scales = (const float*)d_in[1];
    const float* bias   = (const float*)d_in[2];
    const float* lora_A = (const float*)d_in[3];
    const float* lora_B = (const float*)d_in[4];
    const int*   qweight= (const int*)d_in[5];
    const int*   qzeros = (const int*)d_in[6];
    // d_in[7] g_idx unused: g_idx[k] == k/128 by construction
    float* out = (float*)d_out;

    float* ws     = (float*)d_ws;
    float* colsum = ws;                       // 8*4096
    float* stats  = colsum + 8 * IN_F;        // 16
    float* Asw    = stats + 16;               // 4096*16
    float* hbuf   = Asw + IN_F * RNK;         // 4096*16   (~0.66 MB total)

    colabs_kernel<<<dim3(IN_F / 256, 8), 256, 0, stream>>>(x, colsum);
    stats_kernel <<<1, 256, 0, stream>>>(colsum, stats);
    asw_kernel   <<<IN_F / 256, 256, 0, stream>>>(colsum, stats, lora_A, Asw);
    h_kernel     <<<M_TOK / 8, 256, 0, stream>>>(x, Asw, hbuf);
    gemm_kernel  <<<dim3(OUT_F / TN, M_TOK / TM), 256, 0, stream>>>(
        x, qweight, qzeros, scales, bias, lora_B, hbuf, out);
}
